// Spatial_Attn_83236466196698
// MI455X (gfx1250) — matrix operand order repaired and checked by reading, not compiled, not run on hardware
//
#include <hip/hip_runtime.h>

typedef __attribute__((ext_vector_type(16))) __bf16 v16bf;
typedef __attribute__((ext_vector_type(8)))  float  v8f;
typedef unsigned short u16;
typedef unsigned int   u32;

#define BB   16
#define C_CH 32
#define HH   32
#define WW   2048
#define KTOT 1024   // C_CH * HH

// round-to-nearest-even f32 -> bf16 bits
__device__ __forceinline__ u16 bfround(float f) {
    u32 u = __float_as_uint(f);
    u32 r = u + 0x7FFFu + ((u >> 16) & 1u);
    return (u16)(r >> 16);
}

union F16 { v16bf v; uint4 q[2]; };

#define WMMA_BF16(A, B, Cacc) \
    __builtin_amdgcn_wmma_f32_16x16x32_bf16(false, (A), false, (B), (short)0, (Cacc), false, false)

// ---------------------------------------------------------------------------
// Kernel 1: fused Q/K/V projection, ping-pong LDS pipeline.
//   Per (batch, 128-wide tile): C[96 x 128] = Wqkv[96 x 1024] * X[1024 x 128]
//   Staging addresses are (thread-fixed base) + constant*j so global loads
//   take the SADDR+IOFFSET path and LDS stores use offset immediates.
//   Stores: Qt,Kt as bf16 [B][W][32] (transposed), V as bf16 [B][32][W].
// ---------------------------------------------------------------------------
__global__ __launch_bounds__(256) void proj_kernel(
    const float* __restrict__ x,
    const float* __restrict__ wq, const float* __restrict__ bq,
    const float* __restrict__ wk, const float* __restrict__ bk,
    const float* __restrict__ wv, const float* __restrict__ bv,
    u16* __restrict__ qt, u16* __restrict__ kt, u16* __restrict__ vmat)
{
    __shared__ u16 ws[2][96 * 64];    // weights chunk [m=96][k=64], k contiguous
    __shared__ u16 xs[2][128 * 72];   // x chunk transposed [n=128][k=64+pad]

    const int b    = blockIdx.y;
    const int w0   = blockIdx.x * 128;
    const int t    = threadIdx.x;
    const int wave = t >> 5;
    const int lane = t & 31;
    const int lo   = (lane < 16);
    const int nl   = lane & 15;

    // thread-fixed staging coordinates
    const int kfix = t & 63;          // weight k within chunk
    const int mrow = t >> 6;          // weight m base (0..3), +4 per j
    const int n4   = (t & 31) << 2;   // x column group (fixed)
    const int kxr  = t >> 5;          // x row base (0..7), +8 per j

    const float* xg = x + (size_t)b * KTOT * WW + w0 + n4;

    v8f acc[6];
    #pragma unroll
    for (int i = 0; i < 6; i++) acc[i] = {};

    auto stage = [&](u16* __restrict__ wsd, u16* __restrict__ xsd, int k0) {
        // weights: three fixed-pointer loops, 8 elements each
        #pragma unroll
        for (int j = 0; j < 8; ++j) {
            int m = mrow + 4 * j;
            wsd[m * 64 + kfix] = bfround(wq[(size_t)m * KTOT + k0 + kfix]);
        }
        #pragma unroll
        for (int j = 0; j < 8; ++j) {
            int m = mrow + 4 * j;
            wsd[(m + 32) * 64 + kfix] = bfround(wk[(size_t)m * KTOT + k0 + kfix]);
        }
        #pragma unroll
        for (int j = 0; j < 8; ++j) {
            int m = mrow + 4 * j;
            wsd[(m + 64) * 64 + kfix] = bfround(wv[(size_t)m * KTOT + k0 + kfix]);
        }
        // x chunk (64 x 128) transposed -> [n][k]
        #pragma unroll
        for (int j = 0; j < 8; ++j) {
            int krow = kxr + 8 * j;
            const float4 xv = *(const float4*)&xg[(size_t)(k0 + krow) * WW];
            xsd[(n4 + 0) * 72 + krow] = bfround(xv.x);
            xsd[(n4 + 1) * 72 + krow] = bfround(xv.y);
            xsd[(n4 + 2) * 72 + krow] = bfround(xv.z);
            xsd[(n4 + 3) * 72 + krow] = bfround(xv.w);
        }
    };

    stage(&ws[0][0], &xs[0][0], 0);
    __syncthreads();

    const int nB  = (wave << 4) + nl;
    const int kbB = lo ? 0 : 16;
    const int kbA = lo ? 0 : 16;

    for (int c = 0; c < KTOT / 64; ++c) {
        if (c + 1 < KTOT / 64)
            stage(&ws[(c + 1) & 1][0], &xs[(c + 1) & 1][0], (c + 1) * 64);
        const u16* wsb = &ws[c & 1][0];
        const u16* xsb = &xs[c & 1][0];

        #pragma unroll
        for (int ks = 0; ks < 64; ks += 32) {
            F16 fb;   // B fragment: x [32k x 16n]
            {
                const u16* p = &xsb[nB * 72 + ks + kbB];
                fb.q[0] = *(const uint4*)p;
                fb.q[1] = *(const uint4*)(p + 8);
            }
            #pragma unroll
            for (int mt = 0; mt < 6; mt++) {
                F16 fa;   // A fragment: weights [16m x 32k]
                const u16* p = &wsb[(mt * 16 + nl) * 64 + ks + kbA];
                fa.q[0] = *(const uint4*)p;
                fa.q[1] = *(const uint4*)(p + 8);
                acc[mt] = WMMA_BF16(fa.v, fb.v, acc[mt]);
            }
        }
        __syncthreads();
    }

    // --- epilogue: bias add + bf16 store ---
    const int w = w0 + (wave << 4) + nl;

    // Q, K: transposed [B][W][32] -> per lane 8 consecutive channels = 1x b128
    #pragma unroll
    for (int mt = 0; mt < 4; mt++) {
        const float* bias = (mt < 2) ? bq : bk;
        u16*         dst  = (mt < 2) ? qt : kt;
        int cbase = ((mt & 1) << 4) + (lo ? 0 : 8);
        u32 pk[4];
        #pragma unroll
        for (int r = 0; r < 8; r += 2) {
            float v0 = acc[mt][r]     + bias[cbase + r];
            float v1 = acc[mt][r + 1] + bias[cbase + r + 1];
            pk[r >> 1] = (u32)bfround(v0) | ((u32)bfround(v1) << 16);
        }
        uint4 st; st.x = pk[0]; st.y = pk[1]; st.z = pk[2]; st.w = pk[3];
        *(uint4*)&dst[((size_t)(b * WW + w)) * 32 + cbase] = st;
    }
    // V: natural [B][32][W]
    #pragma unroll
    for (int mt = 4; mt < 6; mt++) {
        int cb = ((mt - 4) << 4) + (lo ? 0 : 8);
        #pragma unroll
        for (int r = 0; r < 8; r++) {
            int c = cb + r;
            vmat[((size_t)(b * 32 + c)) * WW + w] = bfround(acc[mt][r] + bv[c]);
        }
    }
}

// ---------------------------------------------------------------------------
// Kernel 2: flash attention (online softmax) + H-broadcast + residual.
//   Each wave: 16 queries x 2048 keys in 64-key chunks (K/V stay L2-hot).
//   Fragments are consumed immediately after their b128 loads to keep
//   register pressure low.
// ---------------------------------------------------------------------------
__global__ __launch_bounds__(256) void attn_kernel(
    const float* __restrict__ x,
    const u16* __restrict__ qt, const u16* __restrict__ kt,
    const u16* __restrict__ vmat,
    float* __restrict__ out)
{
    __shared__ u16   pls[8 * 16 * 64];   // per-wave P transpose staging (16 KB)
    __shared__ float os[128 * 33];       // O tile for coalesced epilogue (~17 KB)

    const int b    = blockIdx.y;
    const int w0   = blockIdx.x * 128;
    const int t    = threadIdx.x;
    const int wave = t >> 5;
    const int lane = t & 31;
    const int lo   = (lane < 16);
    const int nl   = lane & 15;
    const int qb   = w0 + (wave << 4);

    // Q A-fragment: rows = queries, K = 32 channels (contiguous in Qt)
    F16 aq;
    {
        const u16* p = &qt[((size_t)(b * WW + qb + nl)) * 32 + (lo ? 0 : 16)];
        aq.q[0] = *(const uint4*)p;
        aq.q[1] = *(const uint4*)(p + 8);
    }

    v8f o0 = {}, o1 = {};
    float rmax[8], rsum[8];
    #pragma unroll
    for (int r = 0; r < 8; r++) { rmax[r] = -__builtin_inff(); rsum[r] = 0.f; }

    u16* pw = &pls[wave * 1024];         // [16 q][64 kv]
    const v8f zc = {};

    const u16* ktb = &kt[((size_t)(b * WW + nl)) * 32 + (lo ? 0 : 16)];
    const u16* vb0 = &vmat[((size_t)(b * 32 +      nl)) * WW + (lo ? 0 : 16)];
    const u16* vb1 = &vmat[((size_t)(b * 32 + 16 + nl)) * WW + (lo ? 0 : 16)];

    for (int kc = 0; kc < WW; kc += 64) {
        if (kc + 64 < WW) {
            __builtin_prefetch(ktb + (size_t)(kc + 64) * 32, 0, 1);
            __builtin_prefetch(vb0 + kc + 64, 0, 1);
        }
        // --- scores: 4 x 16-key tiles, fragment consumed immediately ---
        v8f s[4];
        {
            const u16* p = ktb + (size_t)kc * 32;
            #pragma unroll
            for (int j = 0; j < 4; j++) {
                F16 bk;
                bk.q[0] = *(const uint4*)(p + j * 512);
                bk.q[1] = *(const uint4*)(p + j * 512 + 8);
                s[j] = WMMA_BF16(aq.v, bk.v, zc);
            }
        }

        // --- online softmax: row r lives in VGPR r (low lanes) / r+8 (high) ---
        #pragma unroll
        for (int r = 0; r < 8; r++) {
            float m = fmaxf(fmaxf(s[0][r], s[1][r]), fmaxf(s[2][r], s[3][r]));
            m = fmaxf(m, __shfl_xor(m, 1));
            m = fmaxf(m, __shfl_xor(m, 2));
            m = fmaxf(m, __shfl_xor(m, 4));
            m = fmaxf(m, __shfl_xor(m, 8));
            float nm = fmaxf(rmax[r], m);
            float sc = __expf(rmax[r] - nm);
            rmax[r] = nm;
            float p0 = __expf(s[0][r] - nm);
            float p1 = __expf(s[1][r] - nm);
            float p2 = __expf(s[2][r] - nm);
            float p3 = __expf(s[3][r] - nm);
            float su = (p0 + p1) + (p2 + p3);
            su += __shfl_xor(su, 1);
            su += __shfl_xor(su, 2);
            su += __shfl_xor(su, 4);
            su += __shfl_xor(su, 8);
            rsum[r] = rsum[r] * sc + su;
            o0[r] *= sc;
            o1[r] *= sc;
            int q = r + (lo ? 0 : 8);
            pw[q * 64 +      nl] = bfround(p0);
            pw[q * 64 + 16 + nl] = bfround(p1);
            pw[q * 64 + 32 + nl] = bfround(p2);
            pw[q * 64 + 48 + nl] = bfround(p3);
        }

        // --- P A-fragments (transpose via per-wave LDS, in-order DS) ---
        F16 pa0, pa1;
        {
            const u16* p = &pw[nl * 64 + (lo ? 0 : 16)];
            pa0.q[0] = *(const uint4*)p;
            pa0.q[1] = *(const uint4*)(p + 8);
            pa1.q[0] = *(const uint4*)(p + 32);
            pa1.q[1] = *(const uint4*)(p + 40);
        }
        // --- V B-fragments consumed immediately ---
        {
            const u16* p0 = vb0 + kc;
            const u16* p1 = vb1 + kc;
            F16 bv;
            bv.q[0] = *(const uint4*)p0;        bv.q[1] = *(const uint4*)(p0 + 8);
            o0 = WMMA_BF16(pa0.v, bv.v, o0);
            bv.q[0] = *(const uint4*)p1;        bv.q[1] = *(const uint4*)(p1 + 8);
            o1 = WMMA_BF16(pa0.v, bv.v, o1);
            bv.q[0] = *(const uint4*)(p0 + 32); bv.q[1] = *(const uint4*)(p0 + 40);
            o0 = WMMA_BF16(pa1.v, bv.v, o0);
            bv.q[0] = *(const uint4*)(p1 + 32); bv.q[1] = *(const uint4*)(p1 + 40);
            o1 = WMMA_BF16(pa1.v, bv.v, o1);
        }
    }

    // normalize and park O tile in LDS (padded stride kills bank conflicts)
    #pragma unroll
    for (int r = 0; r < 8; r++) {
        float inv = 1.0f / rsum[r];
        int q = (wave << 4) + r + (lo ? 0 : 8);
        os[q * 33 + nl]      = o0[r] * inv;
        os[q * 33 + 16 + nl] = o1[r] * inv;
    }
    __syncthreads();

    // broadcast over H + residual add, float4-coalesced along W
    for (int i = t; i < 32 * 32 * 32; i += 256) {    // (c,h) rows x 32 float4
        int q4  = i & 31;
        int row = i >> 5;             // c*32 + h
        int c = row >> 5, h = row & 31;
        size_t g = ((size_t)((b * 32 + c) * 32 + h)) * WW + w0 + (q4 << 2);
        float4 xv = *(const float4*)&x[g];
        float4 ov;
        ov.x = xv.x + os[(q4 * 4 + 0) * 33 + c];
        ov.y = xv.y + os[(q4 * 4 + 1) * 33 + c];
        ov.z = xv.z + os[(q4 * 4 + 2) * 33 + c];
        ov.w = xv.w + os[(q4 * 4 + 3) * 33 + c];
        *(float4*)&out[g] = ov;
    }
}

// ---------------------------------------------------------------------------
extern "C" void kernel_launch(void* const* d_in, const int* in_sizes, int n_in,
                              void* d_out, int out_size, void* d_ws, size_t ws_size,
                              hipStream_t stream) {
    (void)in_sizes; (void)n_in; (void)out_size; (void)ws_size;
    const float* x  = (const float*)d_in[0];
    const float* wq = (const float*)d_in[1];
    const float* bq = (const float*)d_in[2];
    const float* wk = (const float*)d_in[3];
    const float* bk = (const float*)d_in[4];
    const float* wv = (const float*)d_in[5];
    const float* bv = (const float*)d_in[6];
    float* out = (float*)d_out;

    u16* qt = (u16*)d_ws;                       // [B][W][32] bf16, 2 MB
    u16* kt = qt + (size_t)BB * WW * 32;        // [B][W][32] bf16, 2 MB
    u16* vv = kt + (size_t)BB * WW * 32;        // [B][32][W] bf16, 2 MB

    dim3 grid(WW / 128, BB);
    proj_kernel<<<grid, 256, 0, stream>>>(x, wq, bq, wk, bk, wv, bv, qt, kt, vv);
    attn_kernel<<<grid, 256, 0, stream>>>(x, qt, kt, vv, out);
}
